// GraphAttentionLayer_7035156431028
// MI455X (gfx1250) — compile-verified, matrix operand order checked by reading, not run on hardware
//
#include <hip/hip_runtime.h>

typedef unsigned short u16;
typedef unsigned int   u32;
typedef __attribute__((ext_vector_type(16))) _Float16 v16h;
typedef __attribute__((ext_vector_type(8)))  float    v8f;
typedef __attribute__((ext_vector_type(4)))  float    f4;
typedef __attribute__((ext_vector_type(4)))  int      i4;
typedef __attribute__((ext_vector_type(4)))  u32      u4;

#define BS     8
#define NN     2048
#define FIN    128
#define FOUT   64
#define ALPHA  0.2f
#define LOG2E  1.44269504088896340736f

// ---- f32 pair -> packed f16 (v_cvt_pk_rtz_f16_f32, single instruction) ------
__device__ __forceinline__ u32 pk2(float lo, float hi) {
#if __has_builtin(__builtin_amdgcn_cvt_pkrtz)
  auto t = __builtin_amdgcn_cvt_pkrtz(lo, hi);
  u32 u; __builtin_memcpy(&u, &t, 4);
  return u;
#else
  union { _Float16 h[2]; u32 u; } c;
  c.h[0] = (_Float16)lo; c.h[1] = (_Float16)hi;
  return c.u;
#endif
}

__device__ __forceinline__ float fast_exp2(float x) {
#if __has_builtin(__builtin_amdgcn_exp2f)
  return __builtin_amdgcn_exp2f(x);
#else
  return __expf(x * 0.6931471805599453f);
#endif
}

union FragH {            // 16 f16 = 8 VGPRs, matches 16-bit A/B fragment layout
  v16h v;
  u32  u[8];
  u4   q[2];
};

// ---------------- Kernel C: WT[f][k] = f16(W[k][f]) --------------------------
__global__ __launch_bounds__(256) void prep_w(const float* __restrict__ W,
                                              u16* __restrict__ WT) {
  for (int e = threadIdx.x; e < FIN * FOUT; e += 256) {
    int k = e >> 6, f = e & 63;
    WT[f * FIN + k] = (u16)pk2(W[e], 0.f);
  }
}

// ---------------- Kernel A: h = x@W (f16 WMMA), hT + s1/s2 -------------------
// s1/s2 are stored pre-scaled by log2(e): softmax probs p = 2^(lrelu(s1L+s2L)).
__global__ __launch_bounds__(32) void gat_h(const float* __restrict__ x,
                                            const float* __restrict__ avec,
                                            const u16*  __restrict__ WT,
                                            u16*  __restrict__ hT,
                                            float* __restrict__ s1,
                                            float* __restrict__ s2) {
  const int b  = blockIdx.x >> 7;            // 128 row-tiles per batch
  const int n0 = (blockIdx.x & 127) << 4;    // 16 rows per tile
  const int lane = threadIdx.x;
  const int m  = lane & 15;
  const int hi = lane >> 4;

  __shared__ float hlds[16 * FOUT];

  v8f c[4] = {};
  const float* xrow = x + ((size_t)(b * NN + n0 + m)) * FIN;

  for (int kc = 0; kc < 4; ++kc) {           // K = 128 in 4 chunks of 32
    const int kb = kc * 32 + hi * 8;
    f4 xa = *(const f4*)(xrow + kb);
    f4 xb = *(const f4*)(xrow + kb + 4);
    f4 xc = *(const f4*)(xrow + kb + 16);
    f4 xd = *(const f4*)(xrow + kb + 20);
    FragH af;
    af.u[0] = pk2(xa.x, xa.y); af.u[1] = pk2(xa.z, xa.w);
    af.u[2] = pk2(xb.x, xb.y); af.u[3] = pk2(xb.z, xb.w);
    af.u[4] = pk2(xc.x, xc.y); af.u[5] = pk2(xc.z, xc.w);
    af.u[6] = pk2(xd.x, xd.y); af.u[7] = pk2(xd.z, xd.w);
#pragma unroll
    for (int t = 0; t < 4; ++t) {            // 4 f-tiles of 16
      FragH bf;
      const u16* wp = WT + (size_t)(t * 16 + m) * FIN + kb;
      bf.q[0] = *(const u4*)(wp);
      bf.q[1] = *(const u4*)(wp + 16);
      c[t] = __builtin_amdgcn_wmma_f32_16x16x32_f16(
          false, af.v, false, bf.v, (short)0, c[t], false, false);
    }
  }

  // hT[b][f][j] f16 : lane holds column f=t*16+m, rows j = n0+hi*8+r (contiguous)
#pragma unroll
  for (int t = 0; t < 4; ++t) {
    u4 p;
    p.x = pk2(c[t][0], c[t][1]);
    p.y = pk2(c[t][2], c[t][3]);
    p.z = pk2(c[t][4], c[t][5]);
    p.w = pk2(c[t][6], c[t][7]);
    u16* hp = hT + ((size_t)(b * FOUT + t * 16 + m)) * NN + n0 + hi * 8;
    *(u4*)hp = p;
  }

  // stage f32 h tile to LDS, then compute s1 (lanes 0-15) / s2 (lanes 16-31)
#pragma unroll
  for (int t = 0; t < 4; ++t)
#pragma unroll
    for (int r = 0; r < 8; ++r)
      hlds[(hi * 8 + r) * FOUT + t * 16 + m] = c[t][r];
  __syncthreads();

  const float* ap = avec + hi * FOUT;        // a1 or a2
  float s = 0.f;
  for (int f = 0; f < FOUT; ++f) s += hlds[m * FOUT + f] * ap[f];
  float* sp = hi ? s2 : s1;
  sp[b * NN + n0 + m] = s * LOG2E;           // pre-scaled for exp2
}

// ------------- Kernel B: fused masked softmax + att@h + ELU ------------------
// p = adj>0 ? 2^(max(e, 0.2e)) : 0   with e = s1L + s2L (already * log2e).
// Softmax max-subtraction cancels; logits here are O(5) so exp2 cannot overflow.
// Denominator l is accumulated by a 5th WMMA against an all-ones B matrix:
// c[4][r] = sum_j p[row][j], already broadcast across the C-fragment row.
__device__ __forceinline__ float pcalc(int adjv, float s2L, float s1L) {
  float e  = s1L + s2L;
  float lr = fmaxf(e, ALPHA * e);            // leaky-relu (alpha<1)
  float p  = fast_exp2(lr);
  return adjv > 0 ? p : 0.f;
}

__global__ __launch_bounds__(32) void gat_attn(const float* __restrict__ s1g,
                                               const float* __restrict__ s2g,
                                               const int*  __restrict__ adj,
                                               const u16*  __restrict__ hT,
                                               float* __restrict__ out) {
  const int b  = blockIdx.x >> 7;
  const int i0 = (blockIdx.x & 127) << 4;
  const int lane = threadIdx.x;
  const int m  = lane & 15;
  const int hi = lane >> 4;

  const float s1L  = s1g[b * NN + i0 + m];
  const float* s2b = s2g + b * NN;
  const int* adjrow = adj + (size_t)(i0 + m) * NN;

  FragH ones;                                // B = all 1.0h: row-sum reducer
#pragma unroll
  for (int i = 0; i < 8; ++i) ones.u[i] = 0x3C003C00u;

  v8f c[5] = {};                             // 4 f-tiles + denominator

  for (int ch = 0; ch < 64; ++ch) {          // K = 2048 in chunks of 32
    const int j0 = ch * 32;
    FragH af;
#pragma unroll
    for (int g = 0; g < 2; ++g) {
      const int jb = j0 + g * 16 + hi * 8;
      i4 a0 = *(const i4*)(adjrow + jb);
      i4 a1 = *(const i4*)(adjrow + jb + 4);
      f4 s0 = *(const f4*)(s2b + jb);
      f4 s1_ = *(const f4*)(s2b + jb + 4);
      float p0 = pcalc(a0.x, s0.x, s1L);
      float p1 = pcalc(a0.y, s0.y, s1L);
      float p2 = pcalc(a0.z, s0.z, s1L);
      float p3 = pcalc(a0.w, s0.w, s1L);
      float p4 = pcalc(a1.x, s1_.x, s1L);
      float p5 = pcalc(a1.y, s1_.y, s1L);
      float p6 = pcalc(a1.z, s1_.z, s1L);
      float p7 = pcalc(a1.w, s1_.w, s1L);
      af.u[g * 4 + 0] = pk2(p0, p1);
      af.u[g * 4 + 1] = pk2(p2, p3);
      af.u[g * 4 + 2] = pk2(p4, p5);
      af.u[g * 4 + 3] = pk2(p6, p7);
    }
#pragma unroll
    for (int t = 0; t < 4; ++t) {
      FragH bf;
      const u16* hp = hT + ((size_t)(b * FOUT + t * 16 + m)) * NN + j0 + hi * 8;
      bf.q[0] = *(const u4*)(hp);
      bf.q[1] = *(const u4*)(hp + 16);
      c[t] = __builtin_amdgcn_wmma_f32_16x16x32_f16(
          false, af.v, false, bf.v, (short)0, c[t], false, false);
    }
    c[4] = __builtin_amdgcn_wmma_f32_16x16x32_f16(
        false, af.v, false, ones.v, (short)0, c[4], false, false);
  }

  // per-row reciprocal denominator straight from the ones-WMMA accumulator
  float rinv[8];
#pragma unroll
  for (int r = 0; r < 8; ++r)
    rinv[r] = 1.0f / fmaxf(c[4][r], 1e-30f);

  float* orow = out + ((size_t)(b * NN + i0 + hi * 8)) * FOUT + m;
#pragma unroll
  for (int t = 0; t < 4; ++t)
#pragma unroll
    for (int r = 0; r < 8; ++r) {
      float v = c[t][r] * rinv[r];
      v = v > 0.f ? v : fast_exp2(v * LOG2E) - 1.f;   // ELU(alpha=1)
      orow[(size_t)r * FOUT + t * 16] = v;
    }
}

extern "C" void kernel_launch(void* const* d_in, const int* in_sizes, int n_in,
                              void* d_out, int out_size, void* d_ws, size_t ws_size,
                              hipStream_t stream) {
  const float* x   = (const float*)d_in[0];   // (8,2048,128) f32
  const int*   adj = (const int*)d_in[1];     // (2048,2048) i32
  const float* W   = (const float*)d_in[2];   // (128,64) f32
  const float* a   = (const float*)d_in[3];   // (128,1) f32
  float* out = (float*)d_out;                 // (8,2048,64) f32

  char* ws = (char*)d_ws;
  u16*   WT = (u16*)ws;                                   // 64*128  f16 = 16 KB
  u16*   hT = (u16*)(ws + 16384);                         // 8*64*2048 f16 = 2 MB
  float* s1 = (float*)(ws + 16384 + 2097152);             // 8*2048 f32
  float* s2 = s1 + BS * NN;                               // 8*2048 f32

  prep_w  <<<1,    256, 0, stream>>>(W, WT);
  gat_h   <<<BS * (NN / 16), 32, 0, stream>>>(x, a, WT, hT, s1, s2);
  gat_attn<<<BS * (NN / 16), 32, 0, stream>>>(s1, s2, adj, hT, out);
}
